// Pair_Construct_Navie_6451040878734
// MI455X (gfx1250) — matrix operand order repaired and checked by reading, not compiled, not run on hardware
//
#include <hip/hip_runtime.h>
#include <hip/hip_bf16.h>

// Problem constants (from reference: B=32, S=128, D=256, P=64, K=8)
#define BATCH   32
#define SEQ     128
#define KWIN    8
#define NPAIR   2104          // sum of window sizes: 128*17 - 72
#define MTILE   32            // pair rows per block
#define NTILES  66            // ceil(2104/32)
#define DHID    256
#define POSD    64
#define FIN     576           // 2*256 + 64
#define KSTEPS  (FIN / 32)    // 18 WMMA k-steps
#define ASTR    584           // LDS f16 row stride (16B-aligned, conflict pad)
#define HSTR    260           // LDS f32 row stride for h tile
#define LN_EPS  1e-5f

typedef _Float16 v16h __attribute__((ext_vector_type(16)));
typedef _Float16 v8h  __attribute__((ext_vector_type(8)));
typedef float    v8f  __attribute__((ext_vector_type(8)));

// ---------------------------------------------------------------------------
// Kernel 1: tiny prep. (a) collapse kernel@rpe to a 17x64 table (f16);
// (b) enumerate windowed pairs -> ii/jj index arrays + emo_cau_pos output.
// ---------------------------------------------------------------------------
__global__ void prep_kernel(const float* __restrict__ pos_emb,
                            float* __restrict__ out_pairs,   // d_out + B*NPAIR
                            int* __restrict__ ii, int* __restrict__ jj,
                            _Float16* __restrict__ rpe16) {
  const int tid = threadIdx.x;  // 256 threads, 1 block

  // (a) kernel@rpe collapses by rel value:
  // table[r][p] = sum_{r2} (S-|r2-K|) * exp(-(r-r2)^2) * pos_emb[r2][p]
  for (int idx = tid; idx < 17 * POSD; idx += 256) {
    const int r = idx / POSD, p = idx % POSD;
    float acc = 0.f;
    for (int r2 = 0; r2 < 17; ++r2) {
      const int arel = (r2 >= KWIN) ? (r2 - KWIN) : (KWIN - r2);
      const float cnt = (float)(SEQ - arel);
      const float d = (float)(r - r2);
      acc += cnt * expf(-d * d) * pos_emb[r2 * POSD + p];
    }
    rpe16[r * POSD + p] = (_Float16)acc;
  }

  // (b) pair enumeration: thread i handles emotion row i
  if (tid < SEQ) {
    const int i = tid;
    int start = 0;
    for (int q = 0; q < i; ++q) {
      const int qmin = (q - KWIN < 0) ? 0 : q - KWIN;
      const int qmax = (q + KWIN > SEQ - 1) ? SEQ - 1 : q + KWIN;
      start += qmax - qmin + 1;
    }
    const int jmin = (i - KWIN < 0) ? 0 : i - KWIN;
    const int jmax = (i + KWIN > SEQ - 1) ? SEQ - 1 : i + KWIN;
    for (int j = jmin; j <= jmax; ++j) {
      const int t = start + (j - jmin);
      ii[t] = i;
      jj[t] = j;
      out_pairs[2 * t]     = (float)(i + 1);
      out_pairs[2 * t + 1] = (float)(j + 1);
    }
  }
}

// ---------------------------------------------------------------------------
// Kernel 2: W1 f32 -> f16 (once; W1 is reused 67k times in the GEMM)
// ---------------------------------------------------------------------------
__global__ void cvt_w1_kernel(const float* __restrict__ W1,
                              _Float16* __restrict__ W1h) {
  const int idx = blockIdx.x * blockDim.x + threadIdx.x;
  if (idx < DHID * FIN) W1h[idx] = (_Float16)W1[idx];
}

// ---------------------------------------------------------------------------
// Kernel 3: fused pair-gather + GEMM (WMMA f16->f32) + bias + LayerNorm +
// ELU + W2 projection. Grid: (M-tile of 32 pairs, batch). 512 threads =
// 16 wave32s; wave w owns output columns [16w,16w+16) and computes two
// 16x16 WMMA tiles (M rows 0-15 and 16-31) sharing each B fragment.
// ---------------------------------------------------------------------------
__global__ void __launch_bounds__(512)
pair_gemm_kernel(const float* __restrict__ H, const _Float16* __restrict__ W1h,
                 const float* __restrict__ b1, const float* __restrict__ ln_g,
                 const float* __restrict__ ln_b, const float* __restrict__ W2,
                 const float* __restrict__ b2, const int* __restrict__ ii,
                 const int* __restrict__ jj, const _Float16* __restrict__ rpe16,
                 float* __restrict__ scores) {
  __shared__ __align__(16) _Float16 Alds[MTILE * ASTR];  // 32 rows x 576 f16
  __shared__ __align__(16) float    Hlds[MTILE * HSTR];  // 32 rows x 256 f32

  const int b    = blockIdx.y;           // 0..31
  const int tid  = threadIdx.x;
  const int lane = tid & 31;
  const int wave = tid >> 5;             // 0..15
  const int t0   = blockIdx.x * MTILE;

  // ---- stage A tile: [H_i ; H_j ; rpe(rel)] as f16, 16B chunks ----
  // 32 rows x 72 chunks of 8 f16; chunks never straddle segment boundaries.
  for (int c = tid; c < MTILE * 72; c += 512) {
    const int row = c / 72;
    const int cc  = c % 72;
    int t = t0 + row;
    if (t >= NPAIR) t = NPAIR - 1;       // clamp; extra rows never emitted
    v8h v;
    if (cc < 64) {
      const int s = (cc < 32) ? ii[t] : jj[t];
      const float4* src = (const float4*)(H + ((size_t)b * SEQ + s) * DHID +
                                          (size_t)(cc & 31) * 8);
      const float4 f0 = src[0], f1 = src[1];
      v[0] = (_Float16)f0.x; v[1] = (_Float16)f0.y;
      v[2] = (_Float16)f0.z; v[3] = (_Float16)f0.w;
      v[4] = (_Float16)f1.x; v[5] = (_Float16)f1.y;
      v[6] = (_Float16)f1.z; v[7] = (_Float16)f1.w;
    } else {
      const _Float16* rp = rpe16 + (size_t)(jj[t] - ii[t] + KWIN) * POSD;
      v = *(const v8h*)(rp + (cc - 64) * 8);
    }
    *(v8h*)(Alds + row * ASTR + cc * 8) = v;
  }
  __syncthreads();

  // ---- WMMA: 2x(16 M) x 16(N per wave) x 576(K), f32 accumulate ----
  const int nn   = lane & 15;            // A: row M; B: col N (same lane bits)
  const int half = lane >> 4;            // K-half selector per ISA layout
  const int d0   = wave * 16;
  const _Float16* wrow  = W1h + (size_t)(d0 + nn) * FIN;  // B[k][n] = W1[n][k]
  const _Float16* arow0 = Alds + nn * ASTR;
  const _Float16* arow1 = arow0 + 16 * ASTR;

  v8f acc0 = {}, acc1 = {};
#pragma unroll
  for (int ks = 0; ks < KSTEPS; ++ks) {
    const int k0 = ks * 32;
    // B fragment: lane(N=nn, half) holds K = k0 + (e>>3)*16 + half*8 + (e&7)
    v8h blo = *(const v8h*)(wrow + k0 + half * 8);
    v8h bhi = *(const v8h*)(wrow + k0 + 16 + half * 8);
    v16h bfrag = __builtin_shufflevector(blo, bhi, 0, 1, 2, 3, 4, 5, 6, 7,
                                         8, 9, 10, 11, 12, 13, 14, 15);
    // A fragments: lane(M, half) -> K = k0 + (e>>3)*16 + half*8 + (e&7)
    v8h a0lo = *(const v8h*)(arow0 + k0 + half * 8);
    v8h a0hi = *(const v8h*)(arow0 + k0 + 16 + half * 8);
    v16h af0 = __builtin_shufflevector(a0lo, a0hi, 0, 1, 2, 3, 4, 5, 6, 7,
                                       8, 9, 10, 11, 12, 13, 14, 15);
    acc0 = __builtin_amdgcn_wmma_f32_16x16x32_f16(
        false, af0, false, bfrag, (short)0, acc0, false, false);
    v8h a1lo = *(const v8h*)(arow1 + k0 + half * 8);
    v8h a1hi = *(const v8h*)(arow1 + k0 + 16 + half * 8);
    v16h af1 = __builtin_shufflevector(a1lo, a1hi, 0, 1, 2, 3, 4, 5, 6, 7,
                                       8, 9, 10, 11, 12, 13, 14, 15);
    acc1 = __builtin_amdgcn_wmma_f32_16x16x32_f16(
        false, af1, false, bfrag, (short)0, acc1, false, false);
  }

  // ---- bias + scatter f32 tiles to LDS (C layout: VGPR r -> M=r+8*half) ----
  const float bias = b1[d0 + nn];
#pragma unroll
  for (int r = 0; r < 8; ++r) {
    Hlds[(r + 8 * half) * HSTR + d0 + nn]        = acc0[r] + bias;
    Hlds[(16 + r + 8 * half) * HSTR + d0 + nn]   = acc1[r] + bias;
  }
  __syncthreads();

  // ---- LayerNorm + ELU + W2 dot: wave w handles rows w and w+16 ----
  const int col0 = lane * 8;
#pragma unroll
  for (int r2 = 0; r2 < 2; ++r2) {
    const int row = wave + 16 * r2;
    const float* hrow = Hlds + row * HSTR;
    float x[8], s1 = 0.f, s2 = 0.f;
#pragma unroll
    for (int q = 0; q < 8; ++q) {
      x[q] = hrow[col0 + q];
      s1 += x[q];
      s2 += x[q] * x[q];
    }
#pragma unroll
    for (int m = 16; m >= 1; m >>= 1) {
      s1 += __shfl_xor(s1, m, 32);
      s2 += __shfl_xor(s2, m, 32);
    }
    const float mu  = s1 * (1.0f / DHID);
    const float var = s2 * (1.0f / DHID) - mu * mu;
    const float inv = rsqrtf(var + LN_EPS);

    float dot = 0.f;
#pragma unroll
    for (int q = 0; q < 8; ++q) {
      const int col = col0 + q;
      float v = (x[q] - mu) * inv * ln_g[col] + ln_b[col];
      v = v > 0.f ? v : expm1f(v);         // ELU (alpha=1)
      dot += v * W2[col];
    }
#pragma unroll
    for (int m = 16; m >= 1; m >>= 1) dot += __shfl_xor(dot, m, 32);

    const int t = t0 + row;
    if (lane == 0 && t < NPAIR) scores[(size_t)b * NPAIR + t] = dot + b2[0];
  }
}

// ---------------------------------------------------------------------------
extern "C" void kernel_launch(void* const* d_in, const int* in_sizes, int n_in,
                              void* d_out, int out_size, void* d_ws,
                              size_t ws_size, hipStream_t stream) {
  (void)in_sizes; (void)n_in; (void)out_size; (void)ws_size;
  const float* H       = (const float*)d_in[0];  // [32,128,256]
  const float* pos_emb = (const float*)d_in[1];  // [17,64]
  const float* W1      = (const float*)d_in[2];  // [256,576]
  const float* b1      = (const float*)d_in[3];  // [256]
  const float* ln_g    = (const float*)d_in[4];  // [256]
  const float* ln_b    = (const float*)d_in[5];  // [256]
  const float* W2      = (const float*)d_in[6];  // [1,256]
  const float* b2      = (const float*)d_in[7];  // [1]
  float* out = (float*)d_out;                    // scores [32*2104] ++ pairs [2104*2]

  char* ws = (char*)d_ws;
  _Float16* rpe16 = (_Float16*)(ws);             // 17*64 f16 (16B aligned)
  int*      iiA   = (int*)(ws + 4096);           // 2104 ints
  int*      jjA   = (int*)(ws + 16384);          // 2104 ints
  _Float16* W1h   = (_Float16*)(ws + 32768);     // 256*576 f16 (32B aligned)

  prep_kernel<<<1, 256, 0, stream>>>(pos_emb, out + (size_t)BATCH * NPAIR,
                                     iiA, jjA, rpe16);
  cvt_w1_kernel<<<(DHID * FIN + 255) / 256, 256, 0, stream>>>(W1, W1h);
  pair_gemm_kernel<<<dim3(NTILES, BATCH), 512, 0, stream>>>(
      H, W1h, b1, ln_g, ln_b, W2, b2, iiA, jjA, rpe16, out);
}
